// DICNN_CRF_17970143167330
// MI455X (gfx1250) — matrix operand order repaired and checked by reading, not compiled, not run on hardware
//
#include <hip/hip_runtime.h>
#include <hip/hip_bf16.h>
#include <stdint.h>

// ---------------------------------------------------------------------------
// CDNA5 / gfx1250 IDCNN tagger forward pass.
//  - all matmul work on v_wmma_f32_16x16x32_bf16, dual-M tiles per wave
//  - conv input strips staged via GLOBAL_LOAD_ASYNC_TO_LDS_B128 when available
//  - LayerNorm/ReLU epilogues fused in LDS
// ---------------------------------------------------------------------------

typedef __attribute__((ext_vector_type(16))) __bf16 v16bf;
typedef __attribute__((ext_vector_type(8)))  float  v8f;

union BF16x16 { uint32_t u[8]; v16bf v; };

#if defined(__has_builtin)
#if __has_builtin(__builtin_amdgcn_global_load_async_to_lds_b128)
#define USE_ASYNC_STAGE 1
#endif
#endif
#ifndef USE_ASYNC_STAGE
#define USE_ASYNC_STAGE 0
#endif

#if USE_ASYNC_STAGE
// async 16B global -> LDS copy (no VGPR round-trip, tracked by ASYNCcnt)
__device__ __forceinline__ void async_cp16(const uint32_t* g, uint32_t* l) {
  typedef int v4i __attribute__((vector_size(16)));
  __builtin_amdgcn_global_load_async_to_lds_b128(
      (__attribute__((address_space(1))) v4i*)g,
      (__attribute__((address_space(3))) v4i*)l,
      0, 0);
}
#endif

__device__ __forceinline__ void wait_async0() {
#if defined(__has_builtin) && __has_builtin(__builtin_amdgcn_s_wait_asynccnt)
  __builtin_amdgcn_s_wait_asynccnt(0);
#else
  asm volatile("s_wait_asynccnt 0" ::: "memory");
#endif
}

__device__ __forceinline__ unsigned short f2bf(float f) {
  union { float f; uint32_t u; } c; c.f = f;
  uint32_t u = c.u;
  uint32_t r = u + 0x7fffu + ((u >> 16) & 1u);   // round-to-nearest-even
  return (unsigned short)(r >> 16);
}
__device__ __forceinline__ float bf2f(unsigned short b) {
  union { uint32_t u; float f; } c; c.u = ((uint32_t)b) << 16;
  return c.f;
}

__device__ __forceinline__ v8f wmma_bf16(v16bf a, v16bf b, v8f c) {
  return __builtin_amdgcn_wmma_f32_16x16x32_bf16(
      /*neg_a=*/false, a, /*neg_b=*/false, b,
      /*c_mod=*/(short)0, c, /*reuse_a=*/false, /*reuse_b=*/false);
}

// ------------------------- weight / input prep -----------------------------

// fp32 (K,N) row-major -> bf16 (N,K): lane-contiguous K for WMMA B operand.
__global__ void wtrans_kernel(const float* __restrict__ W,
                              unsigned short* __restrict__ out, int K, int N) {
  int i = blockIdx.x * blockDim.x + threadIdx.x;
  if (i >= K * N) return;
  int n = i / K, k = i - n * K;
  out[i] = f2bf(W[(size_t)k * N + n]);
}

// conv weights (O=128, I=128, T=3) fp32 -> bf16 [tap][o][i]
__global__ void wconv_kernel(const float* __restrict__ W,
                             unsigned short* __restrict__ out) {
  int i = blockIdx.x * blockDim.x + threadIdx.x;
  if (i >= 3 * 128 * 128) return;
  int tap = i / (128 * 128);
  int rem = i - tap * 128 * 128;
  int o = rem >> 7, ich = rem & 127;
  out[i] = f2bf(W[((size_t)o * 128 + ich) * 3 + tap]);
}

// embedding gather: tokens*(128/2) uint32 writes of packed bf16 pairs
__global__ void embed_kernel(const int* __restrict__ idx,
                             const float* __restrict__ emb,
                             unsigned short* __restrict__ out, int total) {
  int i = blockIdx.x * blockDim.x + threadIdx.x;
  if (i >= total) return;
  int tok = i >> 6, c2 = i & 63;
  const float* src = emb + (size_t)idx[tok] * 128 + (c2 << 1);
  uint32_t v = (uint32_t)f2bf(src[0]) | ((uint32_t)f2bf(src[1]) << 16);
  ((uint32_t*)out)[i] = v;
}

// ------------------------- WMMA GEMM (linear / head1) ----------------------
// Y(T,N) = X(T,K) @ W(K,N) + b.  Wt bf16 (N,K).  8 waves x 8 N-tiles,
// dual M-tiles (32 tokens) per wave so two WMMA chains interleave and the
// weight tile is reused twice per load.
__global__ __launch_bounds__(256) void linear_wmma_kernel(
    const unsigned short* __restrict__ X,
    const unsigned short* __restrict__ Wt,
    const float* __restrict__ bias,
    unsigned short* __restrict__ Y, int K, int N) {
  int t0   = blockIdx.x * 32;
  int lane = threadIdx.x & 31;
  int wv   = threadIdx.x >> 5;
  int n0   = (blockIdx.y * 8 + wv) * 16;
  int m    = lane & 15;          // A row / B,C column within tile
  int kh   = lane >> 4;          // K-half (A,B) / M-half (C)

  v8f acc0 = {}, acc1 = {};
  const uint32_t* arow0 = (const uint32_t*)(X + (size_t)(t0 + m) * K);
  const uint32_t* arow1 = (const uint32_t*)(X + (size_t)(t0 + 16 + m) * K);
  for (int kc = 0; kc < K; kc += 32) {
    BF16x16 a0, a1, b;
    int o0 = (kc + kh * 16) >> 1, o1 = (kc + 8 + kh * 16) >> 1;
#pragma unroll
    for (int j = 0; j < 4; ++j) {
      a0.u[j] = arow0[o0 + j]; a0.u[4 + j] = arow0[o1 + j];
      a1.u[j] = arow1[o0 + j]; a1.u[4 + j] = arow1[o1 + j];
    }
    const uint32_t* brow =
        (const uint32_t*)(Wt + (size_t)(n0 + m) * K + kc + kh * 16);
#pragma unroll
    for (int j = 0; j < 8; ++j) b.u[j] = brow[j];
    acc0 = wmma_bf16(a0.v, b.v, acc0);
    acc1 = wmma_bf16(a1.v, b.v, acc1);
  }
  int n = n0 + m;
  float bv = bias[n];
#pragma unroll
  for (int r = 0; r < 8; ++r) {
    int row = t0 + r + kh * 8;
    Y[(size_t)row * N + n]        = f2bf(acc0[r] + bv);
    Y[(size_t)(row + 16) * N + n] = f2bf(acc1[r] + bv);
  }
}

// ---------------- conv(K=3, dil d) -> relu -> LN [-> relu -> LN] -----------
// One workgroup: 32 tokens x 128 channels.  36-row halo strip staged in LDS
// (async-to-LDS when available); each wave runs 2 accumulator tiles sharing
// the weight operand: 24 WMMA ops in two interleaved chains.
__global__ __launch_bounds__(256) void conv_ln_kernel(
    const unsigned short* __restrict__ X,   // (T,128) bf16
    const unsigned short* __restrict__ Wc,  // (3,128,128) bf16 [tap][o][i]
    const float* __restrict__ bias,
    unsigned short* __restrict__ Y,         // (T,128) bf16
    int dilation, int doubleLN) {
  __shared__ __align__(16) uint32_t xs[36][64];  // rows l0-2 .. l0+33
  __shared__ float yt[32][132];                  // padded vs bank conflicts
  __shared__ float reds[32][8], redss[32][8];
  __shared__ float mn[32], ivr[32];

  int tid = threadIdx.x;
  int t0  = blockIdx.x * 32;          // token tile base (L=1024 | 32)
  int l0  = t0 & 1023;                // position within the sequence

  // ---- stage 36 x 128 bf16 strip: 576 x 16B chunks ----
  for (int idx = tid; idx < 36 * 16; idx += 256) {
    int r = idx >> 4, c4 = (idx & 15) << 2;     // uint32 column base
    int l = l0 - 2 + r;
    if (l >= 0 && l < 1024) {
      const uint32_t* g =
          (const uint32_t*)(X + (size_t)(t0 - l0 + l) * 128) + c4;
#if USE_ASYNC_STAGE
      async_cp16(g, &xs[r][c4]);
#else
      uint4 v = *(const uint4*)g;
      *(uint4*)&xs[r][c4] = v;
#endif
    } else {
      xs[r][c4] = 0; xs[r][c4 + 1] = 0; xs[r][c4 + 2] = 0; xs[r][c4 + 3] = 0;
    }
  }
#if USE_ASYNC_STAGE
  wait_async0();
#endif
  __syncthreads();

  int lane = tid & 31, wv = tid >> 5;
  int m = lane & 15, kh = lane >> 4;
  int n0 = wv * 16;

  v8f acc0 = {}, acc1 = {};
#pragma unroll
  for (int tap = 0; tap < 3; ++tap) {
    int row0 = m + 2 + (tap - 1) * dilation;    // 0..35 in the halo strip
    const unsigned short* wt = Wc + ((size_t)tap * 128 + (n0 + m)) * 128;
#pragma unroll
    for (int ks = 0; ks < 4; ++ks) {
      int kc = ks * 32;
      BF16x16 a0, a1, b;
      int o0 = (kc + kh * 16) >> 1, o1 = (kc + 8 + kh * 16) >> 1;
#pragma unroll
      for (int j = 0; j < 4; ++j) {
        a0.u[j] = xs[row0][o0 + j];      a0.u[4 + j] = xs[row0][o1 + j];
        a1.u[j] = xs[row0 + 16][o0 + j]; a1.u[4 + j] = xs[row0 + 16][o1 + j];
      }
      const uint32_t* brow = (const uint32_t*)(wt + kc + kh * 16);
#pragma unroll
      for (int j = 0; j < 8; ++j) b.u[j] = brow[j];
      acc0 = wmma_bf16(a0.v, b.v, acc0);
      acc1 = wmma_bf16(a1.v, b.v, acc1);
    }
  }

  {   // bias + relu -> LDS strip (C layout: col = lane&15, row = r + 8*kh)
    float bv = bias[n0 + m];
#pragma unroll
    for (int r = 0; r < 8; ++r) {
      float v0 = acc0[r] + bv, v1 = acc1[r] + bv;
      yt[r + kh * 8][n0 + m]      = v0 > 0.f ? v0 : 0.f;
      yt[16 + r + kh * 8][n0 + m] = v1 > 0.f ? v1 : 0.f;
    }
  }
  __syncthreads();

  int t = tid & 31, g = tid >> 5;     // token / channel-group of 16
  int c0 = g * 16;
  for (int pass = 0;; ++pass) {
    float s = 0.f, ss = 0.f;
#pragma unroll
    for (int j = 0; j < 16; ++j) {
      float v = yt[t][c0 + j];
      s += v; ss += v * v;
    }
    reds[t][g] = s; redss[t][g] = ss;
    __syncthreads();
    if (tid < 32) {
      float S = 0.f, SS = 0.f;
#pragma unroll
      for (int j = 0; j < 8; ++j) { S += reds[tid][j]; SS += redss[tid][j]; }
      float mean = S * (1.f / 128.f);
      float var  = SS * (1.f / 128.f) - mean * mean;
      mn[tid] = mean;
      ivr[tid] = rsqrtf(var + 1e-5f);
    }
    __syncthreads();
    float mean = mn[t], iv = ivr[t];
    if (pass == 0 && doubleLN) {      // block tail: relu -> second LN
#pragma unroll
      for (int j = 0; j < 16; ++j) {
        float z = (yt[t][c0 + j] - mean) * iv;
        yt[t][c0 + j] = z > 0.f ? z : 0.f;
      }
      __syncthreads();
      continue;
    }
    uint32_t ow[8];
#pragma unroll
    for (int j = 0; j < 8; ++j) {
      float z0 = (yt[t][c0 + 2 * j]     - mean) * iv;
      float z1 = (yt[t][c0 + 2 * j + 1] - mean) * iv;
      ow[j] = (uint32_t)f2bf(z0) | ((uint32_t)f2bf(z1) << 16);
    }
    uint32_t* dst = (uint32_t*)(Y + ((size_t)(t0 + t) * 128 + c0));
#pragma unroll
    for (int j = 0; j < 8; ++j) dst[j] = ow[j];
    break;
  }
}

// ------------------------- head2: (T,256) @ (256,20) + b -------------------
__global__ void head2_kernel(const unsigned short* __restrict__ H1,
                             const float* __restrict__ W2,
                             const float* __restrict__ b2,
                             float* __restrict__ out, int total) {
  int i = blockIdx.x * blockDim.x + threadIdx.x;
  if (i >= total) return;
  int tok = i / 20, tg = i - tok * 20;
  const unsigned short* h = H1 + (size_t)tok * 256;
  float s = b2[tg];
#pragma unroll 8
  for (int k = 0; k < 256; ++k) s += bf2f(h[k]) * W2[(size_t)k * 20 + tg];
  out[i] = s;
}

// ---------------------------------------------------------------------------

extern "C" void kernel_launch(void* const* d_in, const int* in_sizes, int n_in,
                              void* d_out, int out_size, void* d_ws, size_t ws_size,
                              hipStream_t stream) {
  (void)in_sizes; (void)n_in; (void)out_size; (void)ws_size;
  const int T = 64 * 1024;            // tokens

  const int*   tok  = (const int*)d_in[0];
  const float* emb  = (const float*)d_in[1];
  const float* linW[4], *linB[4];
  for (int i = 0; i < 4; ++i) {
    linW[i] = (const float*)d_in[2 + 2 * i];
    linB[i] = (const float*)d_in[3 + 2 * i];
  }
  // blocks i at 10+6i: wa, ba, wb, bb, wd, bd
  const float* h1W = (const float*)d_in[34];
  const float* h1b = (const float*)d_in[35];
  const float* h2W = (const float*)d_in[36];
  const float* h2b = (const float*)d_in[37];

  char* ws = (char*)d_ws; size_t off = 0;
  auto alloc = [&](size_t bytes) -> void* {
    void* p = ws + off; off += (bytes + 255) & ~(size_t)255; return p;
  };
  unsigned short* bufA = (unsigned short*)alloc((size_t)T * 128 * 2);
  unsigned short* bufB = (unsigned short*)alloc((size_t)T * 128 * 2);
  unsigned short* bufH = (unsigned short*)alloc((size_t)T * 256 * 2);
  unsigned short* wl[4];
  for (int i = 0; i < 4; ++i) wl[i] = (unsigned short*)alloc(128 * 128 * 2);
  unsigned short* wh1 = (unsigned short*)alloc(128 * 256 * 2);
  unsigned short* wc[12];
  for (int i = 0; i < 12; ++i) wc[i] = (unsigned short*)alloc(3 * 128 * 128 * 2);

  // ---- weight prep ----
  for (int i = 0; i < 4; ++i)
    wtrans_kernel<<<(128 * 128 + 255) / 256, 256, 0, stream>>>(linW[i], wl[i], 128, 128);
  wtrans_kernel<<<(128 * 256 + 255) / 256, 256, 0, stream>>>(h1W, wh1, 128, 256);
  for (int b = 0; b < 4; ++b)
    for (int j = 0; j < 3; ++j)   // wa, wb, wd at offsets 0,2,4 within block
      wconv_kernel<<<(3 * 128 * 128 + 255) / 256, 256, 0, stream>>>(
          (const float*)d_in[10 + 6 * b + 2 * j], wc[b * 3 + j]);

  // ---- embedding ----
  embed_kernel<<<(T * 64 + 255) / 256, 256, 0, stream>>>(tok, emb, bufA, T * 64);

  // ---- 4 x (linear -> conv/relu/ln x3 [+relu/ln]) ----
  dim3 gLin(T / 32, 1);
  for (int i = 0; i < 4; ++i) {
    linear_wmma_kernel<<<gLin, 256, 0, stream>>>(bufA, wl[i], linB[i], bufB, 128, 128);
    const float* ba = (const float*)d_in[10 + 6 * i + 1];
    const float* bb = (const float*)d_in[10 + 6 * i + 3];
    const float* bd = (const float*)d_in[10 + 6 * i + 5];
    conv_ln_kernel<<<T / 32, 256, 0, stream>>>(bufB, wc[i * 3 + 0], ba, bufA, 1, 0);
    conv_ln_kernel<<<T / 32, 256, 0, stream>>>(bufA, wc[i * 3 + 1], bb, bufB, 1, 0);
    conv_ln_kernel<<<T / 32, 256, 0, stream>>>(bufB, wc[i * 3 + 2], bd, bufA, 2, 1);
  }

  // ---- heads ----
  dim3 gH1(T / 32, 2);
  linear_wmma_kernel<<<gH1, 256, 0, stream>>>(bufA, wh1, h1b, bufH, 128, 256);
  head2_kernel<<<(T * 20 + 255) / 256, 256, 0, stream>>>(bufH, h2W, h2b,
                                                         (float*)d_out, T * 20);
}
